// EdgeDecoder_19061064860126
// MI455X (gfx1250) — matrix operand order repaired and checked by reading, not compiled, not run on hardware
//
#include <hip/hip_runtime.h>

typedef __attribute__((ext_vector_type(16))) _Float16 v16h;
typedef __attribute__((ext_vector_type(8)))  _Float16 v8h;
typedef __attribute__((ext_vector_type(8)))  float    v8f;
typedef __attribute__((ext_vector_type(4)))  float    v4f;

#define HIDDEN   128
#define K2       256          // 2*HIDDEN (concat K dimension)
#define NTILES   8            // 128 / 16 output columns of layer 1
#define KCH      8            // 256 / 32 K-chunks
#define EPB      256          // edges per block: 8 waves * 32
#define FRAG_H   512          // halves per (nt,kc) fragment pair-of-chunks

// DPP controls
#define DPP_QUAD_XOR1   0xB1  // quad_perm(1,0,3,2)
#define DPP_QUAD_XOR2   0x4E  // quad_perm(2,3,0,1)
#define DPP_ROW_HMIRROR 0x141 // lane ^ 7  (== xor4 once quads are uniform)
#define DPP_ROW_MIRROR  0x140 // lane ^ 15 (== xor8 once octets are uniform)

union BFrag { v16h v; v8h h[2]; };

template <int CTRL>
__device__ __forceinline__ float dppAdd(float v) {
    int s = __builtin_amdgcn_update_dpp(0, __float_as_int(v), CTRL, 0xF, 0xF, true);
    return v + __int_as_float(s);
}

// Gather one 16x32 f16 A fragment chunk (ISA A-layout) from a node-row pair.
__device__ __forceinline__ v16h gatherA(const float* prow, const float* pcol,
                                        int kc, int hi) {
    const float* base = ((kc < 4) ? prow : pcol) + (kc & 3) * 32 + hi * 8;
    v4f f0 = *(const v4f*)(base);           // K = kbase+0..3
    v4f f1 = *(const v4f*)(base + 4);       // K = kbase+4..7
    v4f f2 = *(const v4f*)(base + 16);      // K = kbase+16..19
    v4f f3 = *(const v4f*)(base + 20);      // K = kbase+20..23
    v16h a;
    #pragma unroll
    for (int j = 0; j < 4; ++j) {
        a[j]      = (_Float16)f0[j];
        a[4 + j]  = (_Float16)f1[j];
        a[8 + j]  = (_Float16)f2[j];
        a[12 + j] = (_Float16)f3[j];
    }
    return a;
}

__global__ __launch_bounds__(256) void edgedec_wmma_kernel(
    const float* __restrict__ z,    // [N_AUTHORS, 128]
    const int*   __restrict__ eli,  // [2, E]
    const float* __restrict__ W1,   // [128, 256] row-major (out, in)
    const float* __restrict__ b1,   // [128]
    const float* __restrict__ W2,   // [1, 128]
    const float* __restrict__ b2,   // [1]
    float*       __restrict__ out,  // [E]
    int E, int ntiles_total)
{
    // f16 W1 pre-swizzled into exact WMMA B-fragment order:
    // chunk index = ((nt*8 + kc)*2 + c)*32 + lane ; each chunk = 16 bytes.
    __shared__ __align__(16) _Float16 w1h[NTILES * KCH * 2 * 32 * 8]; // 64 KB

    const int tid = threadIdx.x;

    // ---- cooperative W1 (f32) -> f16 swizzled LDS fill ----
    for (int i = tid; i < NTILES * KCH * 2 * 32; i += 256) {
        int lane = i & 31;
        int c    = (i >> 5) & 1;
        int kc   = (i >> 6) & 7;
        int nt   = (i >> 9) & 7;
        int n    = nt * 16 + (lane & 15);
        int ks   = kc * 32 + ((lane < 16) ? 0 : 8) + c * 16;
        const float* src = W1 + n * K2 + ks;       // 32B aligned
        v4f f0 = *(const v4f*)(src);
        v4f f1 = *(const v4f*)(src + 4);
        v8h h;
        #pragma unroll
        for (int j = 0; j < 4; ++j) { h[j] = (_Float16)f0[j]; h[4 + j] = (_Float16)f1[j]; }
        *(v8h*)&w1h[i * 8] = h;
    }
    __syncthreads();

    const int wave  = tid >> 5;
    const int lane  = tid & 31;
    const int mlane = lane & 15;     // M row within tile / N col within tile
    const int hi    = lane >> 4;     // half-wave select
    const float b2v = b2[0];
    const _Float16* bbase = &w1h[lane * 8];

    // ---- tile-invariant layer-2 constants: preload out of the hot loop ----
    float b1pre[NTILES], w2pre[NTILES];
    #pragma unroll
    for (int nt = 0; nt < NTILES; ++nt) {
        b1pre[nt] = b1[nt * 16 + mlane];
        w2pre[nt] = W2[nt * 16 + mlane];
    }

    #pragma unroll 1
    for (int tile = blockIdx.x; tile < ntiles_total; tile += gridDim.x) {
        const int ebase = tile * EPB + wave * 32;   // 32 edges per wave
        int e0 = ebase + mlane;
        int e1 = ebase + 16 + mlane;
        if (e0 > E - 1) e0 = E - 1;                 // clamp (EXEC stays full)
        if (e1 > E - 1) e1 = E - 1;
        const float* prow0 = z + (long long)eli[e0] * HIDDEN;
        const float* pcol0 = z + (long long)eli[E + e0] * HIDDEN;
        const float* prow1 = z + (long long)eli[e1] * HIDDEN;
        const float* pcol1 = z + (long long)eli[E + e1] * HIDDEN;

        // ---- gather A fragments for both 16-edge M-tiles ----
        v16h afragA[KCH], afragB[KCH];
        #pragma unroll
        for (int kc = 0; kc < KCH; ++kc) afragA[kc] = gatherA(prow0, pcol0, kc, hi);
        #pragma unroll
        for (int kc = 0; kc < KCH; ++kc) afragB[kc] = gatherA(prow1, pcol1, kc, hi);

        float pacc0[8], pacc1[8];
        #pragma unroll
        for (int r = 0; r < 8; ++r) { pacc0[r] = 0.0f; pacc1[r] = 0.0f; }

        // ---- layer 1 GEMM: B double-buffered, each B feeds TWO WMMAs ----
        {
            BFrag bc;                               // prime pipeline: frag 0
            bc.h[0] = *(const v8h*)(bbase);
            bc.h[1] = *(const v8h*)(bbase + 256);
            #pragma unroll 1
            for (int nt = 0; nt < NTILES; ++nt) {
                const int fb = nt * (KCH * FRAG_H); // halves offset of this nt
                v8f acc0 = {0.f, 0.f, 0.f, 0.f, 0.f, 0.f, 0.f, 0.f};
                v8f acc1 = {0.f, 0.f, 0.f, 0.f, 0.f, 0.f, 0.f, 0.f};
                #pragma unroll
                for (int kc = 0; kc < KCH; ++kc) {
                    // prefetch next fragment (contiguous across nt boundaries);
                    // clamp only for the very last step so we stay in-bounds.
                    int nofs = fb + (kc + 1) * FRAG_H;
                    if (kc == KCH - 1 && nt == NTILES - 1) nofs = fb + kc * FRAG_H;
                    BFrag bn;
                    bn.h[0] = *(const v8h*)(bbase + nofs);
                    bn.h[1] = *(const v8h*)(bbase + nofs + 256);
                    acc0 = __builtin_amdgcn_wmma_f32_16x16x32_f16(
                        false, afragA[kc], false, bc.v, (short)0, acc0, false, false);
                    acc1 = __builtin_amdgcn_wmma_f32_16x16x32_f16(
                        false, afragB[kc], false, bc.v, (short)0, acc1, false, false);
                    bc = bn;
                }
                const float b1v = b1pre[nt];
                const float w2v = w2pre[nt];
                #pragma unroll
                for (int r = 0; r < 8; ++r) {
                    float h0 = acc0[r] + b1v;
                    float h1 = acc1[r] + b1v;
                    h0 = h0 > 0.0f ? h0 : 0.0f;
                    h1 = h1 > 0.0f ? h1 : 0.0f;
                    pacc0[r] = fmaf(h0, w2v, pacc0[r]);
                    pacc1[r] = fmaf(h1, w2v, pacc1[r]);
                }
            }
        }

        // ---- reduce over N (16 lanes per half-wave), pure-VALU DPP tree ----
        #pragma unroll
        for (int r = 0; r < 8; ++r) {
            float v0 = pacc0[r];
            v0 = dppAdd<DPP_QUAD_XOR1>(v0);
            v0 = dppAdd<DPP_QUAD_XOR2>(v0);
            v0 = dppAdd<DPP_ROW_HMIRROR>(v0);
            v0 = dppAdd<DPP_ROW_MIRROR>(v0);
            float v1 = pacc1[r];
            v1 = dppAdd<DPP_QUAD_XOR1>(v1);
            v1 = dppAdd<DPP_QUAD_XOR2>(v1);
            v1 = dppAdd<DPP_ROW_HMIRROR>(v1);
            v1 = dppAdd<DPP_ROW_MIRROR>(v1);
            int oe0 = ebase + hi * 8 + r;           // M-tile 0: M = r + 8*hi
            int oe1 = ebase + 16 + hi * 8 + r;      // M-tile 1
            if (mlane == 0) {
                if (oe0 < E) out[oe0] = v0 + b2v;
                if (oe1 < E) out[oe1] = v1 + b2v;
            }
        }
    }
}

extern "C" void kernel_launch(void* const* d_in, const int* in_sizes, int n_in,
                              void* d_out, int out_size, void* d_ws, size_t ws_size,
                              hipStream_t stream) {
    const float* z   = (const float*)d_in[0];
    const int*   eli = (const int*)d_in[1];
    const float* W1  = (const float*)d_in[2];
    const float* b1  = (const float*)d_in[3];
    const float* W2  = (const float*)d_in[4];
    const float* b2  = (const float*)d_in[5];
    float* out = (float*)d_out;

    const int E = in_sizes[1] / 2;
    const int ntiles = (E + EPB - 1) / EPB;
    int grid = ntiles < 2048 ? ntiles : 2048;
    edgedec_wmma_kernel<<<grid, 256, 0, stream>>>(z, eli, W1, b1, W2, b2, out, E, ntiles);
}
